// M02SameVQ_15195594293432
// MI455X (gfx1250) — matrix operand order repaired and checked by reading, not compiled, not run on hardware
//
#include <hip/hip_runtime.h>
#include <hip/hip_bf16.h>
#include <stdint.h>

// ---------------------------------------------------------------------------
// Problem constants (from reference setup_inputs)
// ---------------------------------------------------------------------------
#define B_    16
#define C_    1024
#define T_    1500
#define NBINS 4096
#define HID   512
#define EMB   256
#define MROWS (B_ * T_)      // 24000 rows of (B*T, C)

// GEMM tiling: block = 256 threads = 8 waves. BM=128 (wave w owns rows
// w*16..w*16+15), BN=128 (each wave computes ALL 8 n-tiles of the block so
// the VQ argmin stays wave-local), KC=32 (one bf16 WMMA K-step).
#define BM 128
#define BN 128
#define KC 32
#define LDK (KC + 8)         // LDS row pad: 40 elems = 80B (=5*16B, keeps 16B align)
#define NT (BN / 16)         // n-tiles per wave

// CDNA5 async LDS path (ASYNCcnt-tracked GLOBAL_LOAD_ASYNC_TO_LDS_B128).
#if __has_builtin(__builtin_amdgcn_global_load_async_to_lds_b128)
#define HAS_ASYNC_LDS 1
#define NBUF 2
#else
#define HAS_ASYNC_LDS 0
#define NBUF 1
#endif

typedef __attribute__((ext_vector_type(16))) __bf16 v16bf;
typedef __attribute__((ext_vector_type(8)))  __bf16 v8bf;
typedef __attribute__((ext_vector_type(8)))  float  v8f;
typedef __attribute__((ext_vector_type(4)))  int    v4i;

__device__ __forceinline__ v16bf pack16(v8bf lo, v8bf hi) {
  v16bf r;
#pragma unroll
  for (int i = 0; i < 8; ++i) { r[i] = lo[i]; r[8 + i] = hi[i]; }
  return r;
}

// A-fragment (16x32 bf16, ISA 7.12.2): lanes 0-15 -> row M=l, K {0..7,16..23};
// lanes 16-31 -> row M=l-16, K {8..15,24..31}.
__device__ __forceinline__ v16bf load_a_frag(const __bf16* ap, int half) {
  return pack16(*(const v8bf*)(ap + half * 8),
                *(const v8bf*)(ap + 16 + half * 8));
}
// B-fragment (32x16 bf16): lanes 0-15 -> col N=l, K {0..7,16..23}; lanes 16-31 -> K {8..15,24..31}.
__device__ __forceinline__ v16bf load_b_frag(const __bf16* bp, int half) {
  return pack16(*(const v8bf*)(bp + half * 8),
                *(const v8bf*)(bp + 16 + half * 8));
}

#if HAS_ASYNC_LDS
// Builtin prototype (from hipcc diagnostic): first arg is v4i in addrspace(1)
// ("__device__"), LDS side is v4i in addrspace(3).
typedef __attribute__((address_space(1))) v4i gv4i_t;
typedef __attribute__((address_space(3))) v4i lv4i_t;
__device__ __forceinline__ void async_cp16(const void* g, void* l) {
  __builtin_amdgcn_global_load_async_to_lds_b128(
      (gv4i_t*)g, (lv4i_t*)l, 0, 0);
}
__device__ __forceinline__ void wait_async0() {
#if __has_builtin(__builtin_amdgcn_s_wait_asynccnt)
  __builtin_amdgcn_s_wait_asynccnt(0);
#else
  asm volatile("s_wait_asynccnt 0x0" ::: "memory");
#endif
}
#endif

// One K-step of WMMA work for a wave: 1 A-fragment x 8 B-fragments.
// B-fragments are preloaded in groups of 4 so the LDS loads batch and the
// DS waits don't serialize every WMMA.
__device__ __forceinline__ void wmma_step(const __bf16 (*As)[LDK],
                                          const __bf16 (*Bs)[LDK],
                                          int w, int half, int l16,
                                          v8f acc[NT]) {
  const __bf16* ap = &As[w * 16 + l16][0];
  v16bf af = load_a_frag(ap, half);
#pragma unroll
  for (int g = 0; g < NT; g += 4) {
    v16bf b0 = load_b_frag(&Bs[(g + 0) * 16 + l16][0], half);
    v16bf b1 = load_b_frag(&Bs[(g + 1) * 16 + l16][0], half);
    v16bf b2 = load_b_frag(&Bs[(g + 2) * 16 + l16][0], half);
    v16bf b3 = load_b_frag(&Bs[(g + 3) * 16 + l16][0], half);
    acc[g + 0] = __builtin_amdgcn_wmma_f32_16x16x32_bf16(false, af, false, b0, (short)0, acc[g + 0], false, false);
    acc[g + 1] = __builtin_amdgcn_wmma_f32_16x16x32_bf16(false, af, false, b1, (short)0, acc[g + 1], false, false);
    acc[g + 2] = __builtin_amdgcn_wmma_f32_16x16x32_bf16(false, af, false, b2, (short)0, acc[g + 2], false, false);
    acc[g + 3] = __builtin_amdgcn_wmma_f32_16x16x32_bf16(false, af, false, b3, (short)0, acc[g + 3], false, false);
  }
}

// ---------------------------------------------------------------------------
// centroid fp32 -> bf16 + row squared norms
// ---------------------------------------------------------------------------
__global__ __launch_bounds__(256)
void k_prep_cent(const float* __restrict__ cent, __bf16* __restrict__ cb,
                 float* __restrict__ cnorm) {
  const int row = blockIdx.x;
  const float* src = cent + (size_t)row * C_;
  __bf16* dst = cb + (size_t)row * C_;
  float s = 0.0f;
  for (int c = threadIdx.x; c < C_; c += 256) {
    float v = src[c];
    s += v * v;
    dst[c] = (__bf16)v;
  }
  for (int off = 16; off > 0; off >>= 1) s += __shfl_xor(s, off, 32);
  __shared__ float red[8];
  if ((threadIdx.x & 31) == 0) red[threadIdx.x >> 5] = s;
  __syncthreads();
  if (threadIdx.x == 0) {
    float t = 0.0f;
#pragma unroll
    for (int i = 0; i < 8; ++i) t += red[i];
    cnorm[row] = t;
  }
}

// ---------------------------------------------------------------------------
// feature (B,C,T) fp32 -> (B*T, C) bf16, LDS-tiled transpose
// ---------------------------------------------------------------------------
__global__ __launch_bounds__(256)
void k_feat2bf(const float* __restrict__ feat, __bf16* __restrict__ xb) {
  __shared__ float tile[32][33];
  const int t0 = blockIdx.x * 32, c0 = blockIdx.y * 32, b = blockIdx.z;
  const int tx = threadIdx.x, ty = threadIdx.y;
#pragma unroll
  for (int i = 0; i < 4; ++i) {
    int c = c0 + ty * 4 + i;
    int t = t0 + tx;
    if (t < T_) tile[ty * 4 + i][tx] = feat[((size_t)b * C_ + c) * T_ + t];
  }
  __syncthreads();
#pragma unroll
  for (int i = 0; i < 4; ++i) {
    int t = t0 + ty * 4 + i;
    int c = c0 + tx;
    if (t < T_) xb[((size_t)(b * T_ + t)) * C_ + c] = (__bf16)tile[tx][ty * 4 + i];
  }
}

// ---------------------------------------------------------------------------
// VQ nearest-centroid: WMMA bf16 GEMM (X . C^T) fused with running argmin of
// (|c|^2 - 2 x.c).  Each wave owns 16 rows x the full 128-col chunk, so the
// argmin reduction is wave-local (shfl over the 16 lanes sharing an M row).
// ---------------------------------------------------------------------------
__global__ __launch_bounds__(256)
void k_dist(const __bf16* __restrict__ X, const __bf16* __restrict__ Cb,
            const float* __restrict__ cnorm, int* __restrict__ idx_out, int Mr) {
  __shared__ __bf16 As[NBUF][BM][LDK];
  __shared__ __bf16 Bs[NBUF][BN][LDK];
  const int tid = threadIdx.x;
  const int w = tid >> 5, lane = tid & 31;
  const int half = lane >> 4, l16 = lane & 15;
  const int m0 = blockIdx.x * BM;

  float best[8];
  int bidx[8];
#pragma unroll
  for (int r = 0; r < 8; ++r) { best[r] = 3.4e38f; bidx[r] = 0; }

#if HAS_ASYNC_LDS
  // pre-compute this thread's 2 A-chunks / 2 B-chunks (16B each)
  int crow[2], coff[2];
#pragma unroll
  for (int i = 0; i < 2; ++i) {
    int c = tid + i * 256;
    crow[i] = c >> 2;
    coff[i] = (c & 3) * 8;  // bf16 elements (8 elems = 16B)
  }
#endif

  for (int n0 = 0; n0 < NBINS; n0 += BN) {
    v8f acc[NT];
#pragma unroll
    for (int j = 0; j < NT; ++j)
#pragma unroll
      for (int r = 0; r < 8; ++r) acc[j][r] = 0.0f;

#if HAS_ASYNC_LDS
    const int nk = C_ / KC;
    // prologue: DMA tile k=0 into buffer 0
#pragma unroll
    for (int i = 0; i < 2; ++i) {
      int arow = m0 + crow[i]; if (arow >= Mr) arow = Mr - 1;
      async_cp16(X + (size_t)arow * C_ + coff[i], &As[0][crow[i]][coff[i]]);
      async_cp16(Cb + (size_t)(n0 + crow[i]) * C_ + coff[i], &Bs[0][crow[i]][coff[i]]);
    }
    for (int kt = 0; kt < nk; ++kt) {
      const int sel = kt & 1;
      wait_async0();           // this wave's DMA for buf[sel] has landed
      __syncthreads();         // everyone's DMA landed; prior reads retired
      if (kt + 1 < nk) {       // DMA next tile while we compute (overlap)
        int k1 = (kt + 1) * KC;
#pragma unroll
        for (int i = 0; i < 2; ++i) {
          int arow = m0 + crow[i]; if (arow >= Mr) arow = Mr - 1;
          async_cp16(X + (size_t)arow * C_ + k1 + coff[i], &As[sel ^ 1][crow[i]][coff[i]]);
          async_cp16(Cb + (size_t)(n0 + crow[i]) * C_ + k1 + coff[i], &Bs[sel ^ 1][crow[i]][coff[i]]);
        }
      }
      wmma_step(As[sel], Bs[sel], w, half, l16, acc);
    }
    __syncthreads();
#else
    const int ar = tid >> 1, seg = (tid & 1) * 16;
    for (int k0 = 0; k0 < C_; k0 += KC) {
      int arow = m0 + ar; if (arow >= Mr) arow = Mr - 1;
      *(v16bf*)(&As[0][ar][seg]) = *(const v16bf*)(X + (size_t)arow * C_ + k0 + seg);
      *(v16bf*)(&Bs[0][ar][seg]) = *(const v16bf*)(Cb + (size_t)(n0 + ar) * C_ + k0 + seg);
      __syncthreads();
      wmma_step(As[0], Bs[0], w, half, l16, acc);
      __syncthreads();
    }
#endif

    // fold this 128-bin chunk into the running per-lane minimum
#pragma unroll
    for (int j = 0; j < NT; ++j) {
      int n = n0 + j * 16 + l16;
      float cn = cnorm[n];
#pragma unroll
      for (int r = 0; r < 8; ++r) {
        float s = cn - 2.0f * acc[j][r];
        if (s < best[r]) { best[r] = s; bidx[r] = n; }
      }
    }
  }
  // cross-lane argmin: row M=r lives in lanes 0-15, row M=8+r in lanes 16-31
#pragma unroll
  for (int r = 0; r < 8; ++r) {
    float v = best[r];
    int bi = bidx[r];
    for (int off = 1; off < 16; off <<= 1) {
      float ov = __shfl_xor(v, off, 32);
      int oi = __shfl_xor(bi, off, 32);
      if (ov < v || (ov == v && oi < bi)) { v = ov; bi = oi; }
    }
    if (l16 == 0) {
      int row = m0 + w * 16 + half * 8 + r;
      if (row < Mr) idx_out[row] = bi;
    }
  }
}

// ---------------------------------------------------------------------------
// lin_dec gather into d_out (B,C,T) + residual (feature - lin_dec) -> bf16
// (B*T, C) for the refiner MLP. Tiled like the transpose.
// ---------------------------------------------------------------------------
__global__ __launch_bounds__(256)
void k_residual(const float* __restrict__ feat, const float* __restrict__ cent,
                const int* __restrict__ idx, float* __restrict__ outf,
                __bf16* __restrict__ spk) {
  __shared__ float tile[32][33];
  const int t0 = blockIdx.x * 32, c0 = blockIdx.y * 32, b = blockIdx.z;
  const int tx = threadIdx.x, ty = threadIdx.y;
#pragma unroll
  for (int i = 0; i < 4; ++i) {
    int c = c0 + ty * 4 + i;
    int t = t0 + tx;
    if (t < T_) {
      int row = b * T_ + t;
      float l = cent[(size_t)idx[row] * C_ + c];
      float f = feat[((size_t)b * C_ + c) * T_ + t];
      outf[((size_t)b * C_ + c) * T_ + t] = l;
      tile[ty * 4 + i][tx] = f - l;
    }
  }
  __syncthreads();
#pragma unroll
  for (int i = 0; i < 4; ++i) {
    int t = t0 + ty * 4 + i;
    int c = c0 + tx;
    if (t < T_) spk[((size_t)(b * T_ + t)) * C_ + c] = (__bf16)tile[tx][ty * 4 + i];
  }
}

// ---------------------------------------------------------------------------
// generic fp32 -> bf16 convert (weights)
// ---------------------------------------------------------------------------
__global__ __launch_bounds__(256)
void k_f2bf(const float* __restrict__ s, __bf16* __restrict__ d, int n) {
  int i = blockIdx.x * 256 + threadIdx.x;
  if (i < n) d[i] = (__bf16)s[i];
}

// ---------------------------------------------------------------------------
// Generic bf16 WMMA GEMM: out(M,N) = X(M,K) . W(N,K)^T + bias, fused epilogue.
// mode 0: bf16 out + LeakyReLU(0.01); mode 1: bf16 out linear;
// mode 2: fp32 accumulate into d_out with (B,C,T) transpose (final layer).
// ---------------------------------------------------------------------------
__global__ __launch_bounds__(256)
void k_gemm(const __bf16* __restrict__ X, const __bf16* __restrict__ W,
            const float* __restrict__ bias, float* __restrict__ outf,
            __bf16* __restrict__ outb, int Mr, int N, int K, int mode) {
  __shared__ __bf16 As[NBUF][BM][LDK];
  __shared__ __bf16 Bs[NBUF][BN][LDK];
  const int tid = threadIdx.x;
  const int w = tid >> 5, lane = tid & 31;
  const int half = lane >> 4, l16 = lane & 15;
  const int m0 = blockIdx.y * BM;
  const int n0 = blockIdx.x * BN;

  v8f acc[NT];
#pragma unroll
  for (int j = 0; j < NT; ++j)
#pragma unroll
    for (int r = 0; r < 8; ++r) acc[j][r] = 0.0f;

#if HAS_ASYNC_LDS
  int crow[2], coff[2];
#pragma unroll
  for (int i = 0; i < 2; ++i) {
    int c = tid + i * 256;
    crow[i] = c >> 2;
    coff[i] = (c & 3) * 8;
  }
  const int nk = K / KC;
#pragma unroll
  for (int i = 0; i < 2; ++i) {
    int arow = m0 + crow[i]; if (arow >= Mr) arow = Mr - 1;
    async_cp16(X + (size_t)arow * K + coff[i], &As[0][crow[i]][coff[i]]);
    async_cp16(W + (size_t)(n0 + crow[i]) * K + coff[i], &Bs[0][crow[i]][coff[i]]);
  }
  for (int kt = 0; kt < nk; ++kt) {
    const int sel = kt & 1;
    wait_async0();
    __syncthreads();
    if (kt + 1 < nk) {
      int k1 = (kt + 1) * KC;
#pragma unroll
      for (int i = 0; i < 2; ++i) {
        int arow = m0 + crow[i]; if (arow >= Mr) arow = Mr - 1;
        async_cp16(X + (size_t)arow * K + k1 + coff[i], &As[sel ^ 1][crow[i]][coff[i]]);
        async_cp16(W + (size_t)(n0 + crow[i]) * K + k1 + coff[i], &Bs[sel ^ 1][crow[i]][coff[i]]);
      }
    }
    wmma_step(As[sel], Bs[sel], w, half, l16, acc);
  }
#else
  const int ar = tid >> 1, seg = (tid & 1) * 16;
  for (int k0 = 0; k0 < K; k0 += KC) {
    int arow = m0 + ar; if (arow >= Mr) arow = Mr - 1;
    *(v16bf*)(&As[0][ar][seg]) = *(const v16bf*)(X + (size_t)arow * K + k0 + seg);
    *(v16bf*)(&Bs[0][ar][seg]) = *(const v16bf*)(W + (size_t)(n0 + ar) * K + k0 + seg);
    __syncthreads();
    wmma_step(As[0], Bs[0], w, half, l16, acc);
    __syncthreads();
  }
#endif

  const int mrow_base = m0 + w * 16 + half * 8;
#pragma unroll
  for (int j = 0; j < NT; ++j) {
    int col = n0 + j * 16 + l16;
    float bz = bias[col];
#pragma unroll
    for (int r = 0; r < 8; ++r) {
      int row = mrow_base + r;
      if (row >= Mr) continue;
      float v = acc[j][r] + bz;
      if (mode == 0) v = (v > 0.0f) ? v : 0.01f * v;
      if (mode <= 1) {
        outb[(size_t)row * N + col] = (__bf16)v;
      } else {
        int b = row / T_, t = row - b * T_;
        float* op = outf + (size_t)b * C_ * T_ + (size_t)col * T_ + t;
        *op = *op + v;   // d_out already holds lin_dec
      }
    }
  }
}

// ---------------------------------------------------------------------------
// host launcher
// ---------------------------------------------------------------------------
extern "C" void kernel_launch(void* const* d_in, const int* in_sizes, int n_in,
                              void* d_out, int out_size, void* d_ws, size_t ws_size,
                              hipStream_t stream) {
  const float* feature  = (const float*)d_in[0];
  const float* centroid = (const float*)d_in[1];
  const float* ew0 = (const float*)d_in[2];  const float* eb0 = (const float*)d_in[3];
  const float* ew1 = (const float*)d_in[4];  const float* eb1 = (const float*)d_in[5];
  const float* ew2 = (const float*)d_in[6];  const float* eb2 = (const float*)d_in[7];
  const float* dw0 = (const float*)d_in[8];  const float* db0 = (const float*)d_in[9];
  const float* dw1 = (const float*)d_in[10]; const float* db1 = (const float*)d_in[11];
  const float* dw2 = (const float*)d_in[12]; const float* db2 = (const float*)d_in[13];
  float* outf = (float*)d_out;

  char* ws = (char*)d_ws;
  size_t off = 0;
  auto alloc = [&](size_t bytes) -> void* {
    void* p = ws + off;
    off = (off + bytes + 255) & ~(size_t)255;
    return p;
  };
  __bf16* xb     = (__bf16*)alloc((size_t)MROWS * C_ * 2);   // feature^T / residual
  __bf16* cb     = (__bf16*)alloc((size_t)NBINS * C_ * 2);
  float*  cnorm  = (float*) alloc((size_t)NBINS * 4);
  int*    idx    = (int*)   alloc((size_t)MROWS * 4);
  __bf16* w0b    = (__bf16*)alloc((size_t)HID * C_ * 2);
  __bf16* w1b    = (__bf16*)alloc((size_t)HID * HID * 2);
  __bf16* w2b    = (__bf16*)alloc((size_t)EMB * HID * 2);
  __bf16* dw0b   = (__bf16*)alloc((size_t)HID * EMB * 2);
  __bf16* dw1b   = (__bf16*)alloc((size_t)HID * HID * 2);
  __bf16* dw2b   = (__bf16*)alloc((size_t)C_ * HID * 2);
  __bf16* act0   = (__bf16*)alloc((size_t)MROWS * HID * 2);
  __bf16* act1   = (__bf16*)alloc((size_t)MROWS * HID * 2);
  (void)ws_size; (void)in_sizes; (void)n_in; (void)out_size;

  const int gm = (MROWS + BM - 1) / BM;  // 188
  dim3 tg((T_ + 31) / 32, C_ / 32, B_);

  // 1) centroid bf16 + norms; feature transpose to bf16
  k_prep_cent<<<NBINS, 256, 0, stream>>>(centroid, cb, cnorm);
  k_feat2bf<<<tg, dim3(32, 8), 0, stream>>>(feature, xb);

  // 2) fused GEMM + argmin VQ search
  k_dist<<<gm, 256, 0, stream>>>(xb, cb, cnorm, idx, MROWS);

  // 3) gather lin_dec into d_out, residual -> bf16
  k_residual<<<tg, dim3(32, 8), 0, stream>>>(feature, centroid, idx, outf, xb);

  // 4) weight conversion
  k_f2bf<<<(HID * C_ + 255) / 256, 256, 0, stream>>>(ew0, w0b, HID * C_);
  k_f2bf<<<(HID * HID + 255) / 256, 256, 0, stream>>>(ew1, w1b, HID * HID);
  k_f2bf<<<(EMB * HID + 255) / 256, 256, 0, stream>>>(ew2, w2b, EMB * HID);
  k_f2bf<<<(HID * EMB + 255) / 256, 256, 0, stream>>>(dw0, dw0b, HID * EMB);
  k_f2bf<<<(HID * HID + 255) / 256, 256, 0, stream>>>(dw1, dw1b, HID * HID);
  k_f2bf<<<(C_ * HID + 255) / 256, 256, 0, stream>>>(dw2, dw2b, C_ * HID);

  // 5) refiner MLP: enc(1024->512->512->256), dec(256->512->512->1024)
  k_gemm<<<dim3(HID / BN, gm), 256, 0, stream>>>(xb,   w0b, eb0, nullptr, act0, MROWS, HID, C_,  0);
  k_gemm<<<dim3(HID / BN, gm), 256, 0, stream>>>(act0, w1b, eb1, nullptr, act1, MROWS, HID, HID, 0);
  k_gemm<<<dim3(EMB / BN, gm), 256, 0, stream>>>(act1, w2b, eb2, nullptr, act0, MROWS, EMB, HID, 1);
  k_gemm<<<dim3(HID / BN, gm), 256, 0, stream>>>(act0, dw0b, db0, nullptr, act1, MROWS, HID, EMB, 0);
  k_gemm<<<dim3(HID / BN, gm), 256, 0, stream>>>(act1, dw1b, db1, nullptr, act0, MROWS, HID, HID, 0);
  // final layer: fp32 transpose-add into d_out (which holds lin_dec)
  k_gemm<<<dim3(C_ / BN, gm), 256, 0, stream>>>(act0, dw2b, db2, outf, nullptr, MROWS, C_, HID, 2);
}
